// AutoregressiveDecoder_70093866271176
// MI455X (gfx1250) — matrix operand order repaired and checked by reading, not compiled, not run on hardware
//
#include <hip/hip_runtime.h>

typedef __attribute__((ext_vector_type(16))) _Float16 v16h;
typedef __attribute__((ext_vector_type(8)))  float    v8f;

#define HID    128
#define GATES  384
#define LATD   256
#define OUTD   6
#define STEPS  50
#define NSEQ   (512*128)

// fragged f16 weight offsets in the workspace (units: halfs)
// fragment = 512 halfs = 32 lanes x 16 halfs (one WMMA B operand, 32x16 K-major)
#define O_WLAT  0            // 24 tiles x 8 K-chunks  (W_ih0[:, :256])
#define O_WPREV 98304        // 24 tiles x 1 K-chunk   (W_ih0[:, 256:262] padded to 32)
#define O_WHH0  110592       // 24 x 4
#define O_WIH1  159744       // 24 x 4
#define O_WHH1  208896       // 24 x 4
#define O_WHEAD 258048       // 1  x 4  (rows 0-5 = W_out, row 6 = W_stop, rest 0)
#define W_TOTAL 260096

__device__ __forceinline__ float sigmoid_fast(float x) { return 1.0f / (1.0f + __expf(-x)); }
__device__ __forceinline__ float tanh_fast(float x)    { return 1.0f - 2.0f / (__expf(2.0f * x) + 1.0f); }

__device__ __forceinline__ v8f wmma16(v16h a, v16h b, v8f c) {
    return __builtin_amdgcn_wmma_f32_16x16x32_f16(false, a, false, b, (short)0, c, false, false);
}

// B fragment: lane l holds W[g = t*16 + (l&15)][k0 + (l>>4)*8 + e + (e>=8 ? 8 : 0)], e=0..15 (pre-swizzled)
__device__ __forceinline__ v16h loadB(const _Float16* base, int frag, int lane) {
    return ((const v16h*)base)[frag * 32 + lane];
}

// A fragment (16x32 f16) from LDS row-major buffer: lanes 0-15 hold K=[k0..k0+7],[k0+16..k0+23],
// lanes 16-31 hold K=[k0+8..k0+15],[k0+24..k0+31]  (ISA 7.12.2 layout)
__device__ __forceinline__ v16h loadA_lds(const _Float16* buf, int stride, int lrow, int hlf, int k0) {
    const _Float16* r = buf + lrow * stride + k0 + hlf * 8;
    v16h a;
#pragma unroll
    for (int i = 0; i < 8; i++) { a[i] = r[i]; a[8 + i] = r[16 + i]; }
    return a;
}

__device__ __forceinline__ v8f splat8(float v) {
    v8f c;
#pragma unroll
    for (int j = 0; j < 8; j++) c[j] = v;
    return c;
}

// ---------------- prep: convert + swizzle all weights to fragged f16 ----------------
__global__ void prep_weights(const float* __restrict__ Wih0, const float* __restrict__ Whh0,
                             const float* __restrict__ Wih1, const float* __restrict__ Whh1,
                             const float* __restrict__ Wout, const float* __restrict__ Wstop,
                             _Float16* __restrict__ ws) {
    int i = blockIdx.x * blockDim.x + threadIdx.x;
    if (i >= W_TOTAL) return;
    int l, nc, mode;
    if (i < O_WPREV)      { l = i - O_WLAT;  nc = 8; mode = 0; }
    else if (i < O_WHH0)  { l = i - O_WPREV; nc = 1; mode = 1; }
    else if (i < O_WIH1)  { l = i - O_WHH0;  nc = 4; mode = 2; }
    else if (i < O_WHH1)  { l = i - O_WIH1;  nc = 4; mode = 3; }
    else if (i < O_WHEAD) { l = i - O_WHH1;  nc = 4; mode = 4; }
    else                  { l = i - O_WHEAD; nc = 4; mode = 5; }
    int frag = l >> 9;
    int r    = l & 511;
    int lane = r >> 4;
    int e    = r & 15;
    int t = frag / nc, c = frag % nc;
    int g   = t * 16 + (lane & 15);
    int col = c * 32 + (lane >> 4) * 8 + e + ((e >> 3) << 3);
    float v = 0.0f;
    switch (mode) {
        case 0: v = Wih0[g * 262 + col]; break;
        case 1: v = (col < 6) ? Wih0[g * 262 + 256 + col] : 0.0f; break;
        case 2: v = Whh0[g * HID + col]; break;
        case 3: v = Wih1[g * HID + col]; break;
        case 4: v = Whh1[g * HID + col]; break;
        case 5: v = (g < 6) ? Wout[g * HID + col] : ((g == 6) ? Wstop[col] : 0.0f); break;
    }
    ws[i] = (_Float16)v;
}

// ---------------- main decoder: 16 sequences per block, 8 waves ----------------
__global__ __launch_bounds__(256, 1)
void decoder_kernel(const float* __restrict__ lat, const float* __restrict__ start,
                    const float* __restrict__ b_ih0, const float* __restrict__ b_hh0,
                    const float* __restrict__ b_ih1, const float* __restrict__ b_hh1,
                    const float* __restrict__ b_out, const float* __restrict__ b_stop,
                    const _Float16* __restrict__ ws,
                    float* __restrict__ outStates, float* __restrict__ outStops) {
    __shared__ __align__(16) _Float16 h0h[16 * 136];   // f16 copy of h0 (A fragments), padded stride
    __shared__ __align__(16) _Float16 h1h[16 * 136];
    __shared__ __align__(16) _Float16 prevh[16 * 40];  // prev token padded 6->32 + stride pad
    __shared__ float h0f[16 * 132];                    // f32 h state, padded stride
    __shared__ float h1f[16 * 132];

    const int tid  = threadIdx.x;
    const int wave = tid >> 5, lane = tid & 31;
    const int hlf  = lane >> 4, lrow = lane & 15;
    const int seqBase = blockIdx.x * 16;

    // init shared state
    for (int i = tid; i < 16 * 132; i += 256) { h0f[i] = 0.f; h1f[i] = 0.f; }
    for (int i = tid; i < 16 * 136; i += 256) { h0h[i] = (_Float16)0.f; h1h[i] = (_Float16)0.f; }
    for (int i = tid; i < 16 * 40;  i += 256) {
        int c = i % 40;
        prevh[i] = (c < 6) ? (_Float16)start[c] : (_Float16)0.f;
    }

    const int t0 = wave, t1 = wave + 8, t2 = wave + 16;  // r, z, n gate tiles for hidden cols [16w,16w+16)

    // ---- explicit register residency: all step-invariant weight fragments ----
    v16h wp0 = loadB(ws + O_WPREV, t0, lane);
    v16h wp1 = loadB(ws + O_WPREV, t1, lane);
    v16h wp2 = loadB(ws + O_WPREV, t2, lane);
    v16h whh0[3][4], wih1[3][4], whh1[3][4];
#pragma unroll
    for (int c = 0; c < 4; c++) {
        whh0[0][c] = loadB(ws + O_WHH0, t0 * 4 + c, lane);
        whh0[1][c] = loadB(ws + O_WHH0, t1 * 4 + c, lane);
        whh0[2][c] = loadB(ws + O_WHH0, t2 * 4 + c, lane);
        wih1[0][c] = loadB(ws + O_WIH1, t0 * 4 + c, lane);
        wih1[1][c] = loadB(ws + O_WIH1, t1 * 4 + c, lane);
        wih1[2][c] = loadB(ws + O_WIH1, t2 * 4 + c, lane);
        whh1[0][c] = loadB(ws + O_WHH1, t0 * 4 + c, lane);
        whh1[1][c] = loadB(ws + O_WHH1, t1 * 4 + c, lane);
        whh1[2][c] = loadB(ws + O_WHH1, t2 * 4 + c, lane);
    }
    v16h whead[4];
    if (wave == 0) {
#pragma unroll
        for (int c = 0; c < 4; c++) whead[c] = loadB(ws + O_WHEAD, c, lane);
    }

    // per-step bias scalars (hoisted): r/z gates use merged b_ih + b_hh
    const float bL1r  = b_ih1[t0 * 16 + lrow] + b_hh1[t0 * 16 + lrow];
    const float bL1z  = b_ih1[t1 * 16 + lrow] + b_hh1[t1 * 16 + lrow];
    const float bL1in = b_ih1[t2 * 16 + lrow];
    const float bL1hn = b_hh1[t2 * 16 + lrow];
    const float bL0hn = b_hh0[t2 * 16 + lrow];
    const float bHead = (lrow < 6) ? b_out[lrow] : ((lrow == 6) ? b_stop[0] : 0.0f);

    // --- hoisted step-invariant gi0_lat = lat @ Wlat^T + biases (kept in registers) ---
    // r/z tiles get b_ih0 + b_hh0 baked in (merged-accumulator trick); n tile gets b_ih0 only.
    v8f gl0 = splat8(b_ih0[t0 * 16 + lrow] + b_hh0[t0 * 16 + lrow]);
    v8f gl1 = splat8(b_ih0[t1 * 16 + lrow] + b_hh0[t1 * 16 + lrow]);
    v8f gl2 = splat8(b_ih0[t2 * 16 + lrow]);
#pragma unroll 2
    for (int c = 0; c < 8; c++) {
        const float* r = lat + (size_t)(seqBase + lrow) * LATD + c * 32 + hlf * 8;
        v16h a;
#pragma unroll
        for (int i = 0; i < 8; i++) { a[i] = (_Float16)r[i]; a[8 + i] = (_Float16)r[16 + i]; }
        gl0 = wmma16(a, loadB(ws + O_WLAT, t0 * 8 + c, lane), gl0);
        gl1 = wmma16(a, loadB(ws + O_WLAT, t1 * 8 + c, lane), gl1);
        gl2 = wmma16(a, loadB(ws + O_WLAT, t2 * 8 + c, lane), gl2);
    }
    __syncthreads();

    for (int step = 0; step < STEPS; step++) {
        // ---- layer 0 ----
        // acc_r = (b_ih0+b_hh0) + lat-part + prev-part + h0@Whh0_r   (gi_r + gh_r merged)
        // acc_z likewise; n-gate keeps gi_n / gh_n separate for r*h_n.
        v16h ap = loadA_lds(prevh, 40, lrow, hlf, 0);
        v8f acc_r = wmma16(ap, wp0, gl0);
        v8f acc_z = wmma16(ap, wp1, gl1);
        v8f gi_n  = wmma16(ap, wp2, gl2);
        v8f gh_n  = splat8(bL0hn);
#pragma unroll
        for (int c = 0; c < 4; c++) {
            v16h a = loadA_lds(h0h, 136, lrow, hlf, c * 32);
            acc_r = wmma16(a, whh0[0][c], acc_r);
            acc_z = wmma16(a, whh0[1][c], acc_z);
            gh_n  = wmma16(a, whh0[2][c], gh_n);
        }
        __syncthreads();  // all waves done reading h0h/prevh
        {
            const int col = wave * 16 + lrow;
#pragma unroll
            for (int j = 0; j < 8; j++) {
                float rr = sigmoid_fast(acc_r[j]);
                float zz = sigmoid_fast(acc_z[j]);
                float nn = tanh_fast(gi_n[j] + rr * gh_n[j]);
                int m = j + hlf * 8;
                float hp = h0f[m * 132 + col];
                float hn = (1.0f - zz) * nn + zz * hp;
                h0f[m * 132 + col] = hn;
                h0h[m * 136 + col] = (_Float16)hn;
            }
        }
        __syncthreads();

        // ---- layer 1 ----
        acc_r = splat8(bL1r);
        acc_z = splat8(bL1z);
        gi_n  = splat8(bL1in);
        gh_n  = splat8(bL1hn);
#pragma unroll
        for (int c = 0; c < 4; c++) {
            v16h a0 = loadA_lds(h0h, 136, lrow, hlf, c * 32);
            v16h a1 = loadA_lds(h1h, 136, lrow, hlf, c * 32);
            acc_r = wmma16(a0, wih1[0][c], acc_r);
            acc_r = wmma16(a1, whh1[0][c], acc_r);
            acc_z = wmma16(a0, wih1[1][c], acc_z);
            acc_z = wmma16(a1, whh1[1][c], acc_z);
            gi_n  = wmma16(a0, wih1[2][c], gi_n);
            gh_n  = wmma16(a1, whh1[2][c], gh_n);
        }
        __syncthreads();
        {
            const int col = wave * 16 + lrow;
#pragma unroll
            for (int j = 0; j < 8; j++) {
                float rr = sigmoid_fast(acc_r[j]);
                float zz = sigmoid_fast(acc_z[j]);
                float nn = tanh_fast(gi_n[j] + rr * gh_n[j]);
                int m = j + hlf * 8;
                float hp = h1f[m * 132 + col];
                float hn = (1.0f - zz) * nn + zz * hp;
                h1f[m * 132 + col] = hn;
                h1h[m * 136 + col] = (_Float16)hn;
            }
        }
        __syncthreads();

        // ---- heads (wave 0): state = h1 @ W_out^T + b_out ; stop = sigmoid(h1 @ W_stop^T + b_stop)
        if (wave == 0) {
            v8f hc = splat8(bHead);
#pragma unroll
            for (int c = 0; c < 4; c++)
                hc = wmma16(loadA_lds(h1h, 136, lrow, hlf, c * 32), whead[c], hc);
#pragma unroll
            for (int j = 0; j < 8; j++) {
                int m = j + hlf * 8;
                size_t seq = (size_t)(seqBase + m);
                if (lrow < 6) {
                    outStates[(seq * STEPS + step) * OUTD + lrow] = hc[j];
                    prevh[m * 40 + lrow] = (_Float16)hc[j];   // feedback token
                } else if (lrow == 6) {
                    outStops[seq * STEPS + step] = sigmoid_fast(hc[j]);
                }
            }
        }
        __syncthreads();  // prevh visible to all waves next step
    }
}

extern "C" void kernel_launch(void* const* d_in, const int* in_sizes, int n_in,
                              void* d_out, int out_size, void* d_ws, size_t ws_size,
                              hipStream_t stream) {
    const float* latent = (const float*)d_in[0];
    const float* start  = (const float*)d_in[1];
    const float* Wih0   = (const float*)d_in[2];
    const float* Whh0   = (const float*)d_in[3];
    const float* bih0   = (const float*)d_in[4];
    const float* bhh0   = (const float*)d_in[5];
    const float* Wih1   = (const float*)d_in[6];
    const float* Whh1   = (const float*)d_in[7];
    const float* bih1   = (const float*)d_in[8];
    const float* bhh1   = (const float*)d_in[9];
    const float* Wout   = (const float*)d_in[10];
    const float* bout   = (const float*)d_in[11];
    const float* Wstop  = (const float*)d_in[12];
    const float* bstop  = (const float*)d_in[13];

    _Float16* wsh = (_Float16*)d_ws;
    float* outS = (float*)d_out;
    float* outP = outS + (size_t)NSEQ * STEPS * OUTD;

    prep_weights<<<(W_TOTAL + 255) / 256, 256, 0, stream>>>(Wih0, Whh0, Wih1, Whh1, Wout, Wstop, wsh);
    decoder_kernel<<<NSEQ / 16, 256, 0, stream>>>(latent, start, bih0, bhh0, bih1, bhh1,
                                                  bout, bstop, wsh, outS, outP);
}
